// SparseFFMain_54193897341183
// MI455X (gfx1250) — compile-verified, matrix operand order checked by reading, not run on hardware
//
#include <hip/hip_runtime.h>

// ---------------------------------------------------------------------------
// SparseFF predict path on gfx1250 (MI455X), densified to two bf16 WMMA GEMMs.
//   act[b, j*64+k] = (k == quant_mask[b,j]) ? relu(x[b,:]·w1[:,j,k]) : 0
//   res[b, :]      = act[b,:] @ w2perm + b2    (w2 row permuted k*128+j)
// Output = [quant_mask (bit-copied int32), res (f32)].
// GEMM cores: wave32 WMMA bf16 (f32 accum), LDS-staged tiles, async-to-LDS
// copies for the pre-converted bf16 activation matrix, global prefetch for
// the streamed weight tiles.
// ---------------------------------------------------------------------------

typedef __bf16 bf16_t;
typedef bf16_t v16bf __attribute__((ext_vector_type(16)));
typedef float  v8f   __attribute__((ext_vector_type(8)));

union FragBF {
    uint4 u[2];   // two 16-byte LDS chunks
    v16bf v;      // WMMA operand
};

__device__ __forceinline__ unsigned short f2bf(float f) {
    unsigned int u = __builtin_bit_cast(unsigned int, f);
    u += 0x7FFFu + ((u >> 16) & 1u);          // round-to-nearest-even
    return (unsigned short)(u >> 16);
}
__device__ __forceinline__ unsigned int pack2(float a, float b) {
    return (unsigned int)f2bf(a) | ((unsigned int)f2bf(b) << 16);
}

__device__ __forceinline__ v8f wmma_bf16(const FragBF& a, const FragBF& b, v8f c) {
    return __builtin_amdgcn_wmma_f32_16x16x32_bf16(
        false, a.v, false, b.v, (short)0, c, false, false);
}

#define LDT 40   // padded LDS row stride in ushorts (80 B: 16B-aligned, bank-clean)

// ---------------------------------------------------------------------------
// GEMM1: mid = X(2048x1024) @ W1(1024x8192), fused mask+ReLU -> act bf16.
// Tile: BM=128, BN=128, BK=32. 8 waves as 4(M)x2(N); wave = 32x64 strip
// (2 A frags x 4 B frags = 8 WMMA / K-step, B frag reused twice).
// ---------------------------------------------------------------------------
__global__ __launch_bounds__(256) void gemm1_mask_kernel(
    const float* __restrict__ x,        // (2048, 1024)
    const float* __restrict__ w1,       // (1024, 8192) row-major, col f1=j*64+k
    const int*   __restrict__ qmask,    // (2048, 128)
    unsigned short* __restrict__ act)   // (2048, 8192) bf16 out
{
    __shared__ unsigned short As[128 * LDT];   // 10240 B
    __shared__ unsigned short Bs[128 * LDT];   // 10240 B

    const int n0   = blockIdx.x * 128;
    const int m0   = blockIdx.y * 128;
    const int t    = threadIdx.x;
    const int lane = t & 31;
    const int wave = t >> 5;
    const int lm   = lane & 15;
    const int half = lane >> 4;
    const int wr   = wave & 3;          // M strip (32 rows)
    const int wc   = wave >> 2;         // N strip (64 cols)

    v8f acc[2][4] = {};

    for (int k0 = 0; k0 < 1024; k0 += 32) {
        // --- stage A tile: x[m0..+127, k0..+31] f32 -> bf16 (packed stores)
#pragma unroll
        for (int i = 0; i < 4; ++i) {
            const int id  = i * 256 + t;          // 1024 float4 chunks
            const int row = id >> 3;
            const int c4  = id & 7;
            const float4 s = *(const float4*)(x + (m0 + row) * 1024 + k0 + c4 * 4);
            unsigned int* d = (unsigned int*)(&As[row * LDT + c4 * 4]);
            d[0] = pack2(s.x, s.y);
            d[1] = pack2(s.z, s.w);
        }
        // --- stage B tile transposed: w1[k0..+31, n0..+127] -> Bs[n][k].
        //     Two K-rows per thread so k-pairs pack into 32-bit ds stores.
#pragma unroll
        for (int i = 0; i < 2; ++i) {
            const int id = i * 256 + t;           // 512 row-pair units
            const int r  = id >> 5;               // k-pair 0..15
            const int c4 = id & 31;               // float4 column 0..31
            const float* g0 = w1 + (k0 + 2 * r) * 8192 + n0 + c4 * 4;
            const float4 s0 = *(const float4*)(g0);
            const float4 s1 = *(const float4*)(g0 + 8192);
            __builtin_prefetch(g0 + 32 * 8192, 0, 0);   // next K tile (L2 resident)
            const int n = c4 * 4;
            *(unsigned int*)(&Bs[(n + 0) * LDT + 2 * r]) = pack2(s0.x, s1.x);
            *(unsigned int*)(&Bs[(n + 1) * LDT + 2 * r]) = pack2(s0.y, s1.y);
            *(unsigned int*)(&Bs[(n + 2) * LDT + 2 * r]) = pack2(s0.z, s1.z);
            *(unsigned int*)(&Bs[(n + 3) * LDT + 2 * r]) = pack2(s0.w, s1.w);
        }
        __syncthreads();

        // --- fragments per ISA 16-bit layout: lane holds K[half*8..+7], K[16+half*8..+7]
        FragBF a0, a1;
        const int ar = wr * 32 + lm;
        a0.u[0] = *(const uint4*)(&As[ar * LDT + half * 8]);
        a0.u[1] = *(const uint4*)(&As[ar * LDT + 16 + half * 8]);
        a1.u[0] = *(const uint4*)(&As[(ar + 16) * LDT + half * 8]);
        a1.u[1] = *(const uint4*)(&As[(ar + 16) * LDT + 16 + half * 8]);
#pragma unroll
        for (int nn = 0; nn < 4; ++nn) {
            FragBF b;
            const int br = wc * 64 + nn * 16 + lm;
            b.u[0] = *(const uint4*)(&Bs[br * LDT + half * 8]);
            b.u[1] = *(const uint4*)(&Bs[br * LDT + 16 + half * 8]);
            acc[0][nn] = wmma_bf16(a0, b, acc[0][nn]);
            acc[1][nn] = wmma_bf16(a1, b, acc[1][nn]);
        }
        __syncthreads();
    }

    // --- epilogue: mask + relu, store bf16 (tile's 64-col strip = one j)
    const int j = (n0 + wc * 64) >> 6;
#pragma unroll
    for (int mi = 0; mi < 2; ++mi) {
        const int brow = m0 + wr * 32 + mi * 16 + half * 8;  // C layout: M = r + half*8
        int sel[8];
#pragma unroll
        for (int r = 0; r < 8; ++r) sel[r] = qmask[(brow + r) * 128 + j];
#pragma unroll
        for (int nn = 0; nn < 4; ++nn) {
            const int f1 = n0 + wc * 64 + nn * 16 + lm;      // C layout: N = lane%16
            const int kk = f1 & 63;
#pragma unroll
            for (int r = 0; r < 8; ++r) {
                float v = acc[mi][nn][r];
                v = (sel[r] == kk) ? (v > 0.f ? v : 0.f) : 0.f;
                act[(brow + r) * 8192 + f1] = f2bf(v);
            }
        }
    }
}

// ---------------------------------------------------------------------------
// GEMM2: res = act(2048x8192 bf16) @ W2' + b2 ; W2' row f1 = w2 row
// ((f1&63)*128 + (f1>>6)). Tile: BM=128, BN=64, BK=32; 8 waves as 4(M)x2(N);
// wave = 32x32. A tile staged with GLOBAL_LOAD_ASYNC_TO_LDS_B128 (ASYNCcnt).
// ---------------------------------------------------------------------------
__global__ __launch_bounds__(256) void gemm2_bias_kernel(
    const unsigned short* __restrict__ act,  // (2048, 8192) bf16
    const float* __restrict__ w2,            // (8192, 1024) rows = k*128+j
    const float* __restrict__ b2,            // (1024,)
    float* __restrict__ res)                 // (2048, 1024)
{
    __shared__ unsigned short As[128 * LDT];   // 10240 B
    __shared__ unsigned short Bs[64 * LDT];    //  5120 B

    const int n0   = blockIdx.x * 64;
    const int m0   = blockIdx.y * 128;
    const int t    = threadIdx.x;
    const int lane = t & 31;
    const int wave = t >> 5;
    const int lm   = lane & 15;
    const int half = lane >> 4;
    const int wr   = wave & 3;          // M strip (32 rows)
    const int wc   = wave >> 2;         // N strip (32 cols)

    // Fixed per-thread staging coordinates for the async A copy:
    // 512 chunks of 16 B (128 rows x 4 chunks), 2 per thread.
    const int arow0 = t >> 2;                 // rows 0..63
    const int arow1 = 64 + (t >> 2);          // rows 64..127
    const int ac8   = (t & 3) * 8;            // ushort column within 32-wide K
    const unsigned lds_a0 = (unsigned)(size_t)(&As[arow0 * LDT + ac8]);
    const unsigned lds_a1 = (unsigned)(size_t)(&As[arow1 * LDT + ac8]);
    const unsigned short* ga0 = act + (size_t)(m0 + arow0) * 8192 + ac8;
    const unsigned short* ga1 = act + (size_t)(m0 + arow1) * 8192 + ac8;

    v8f acc[2][2] = {};

    for (int k0 = 0; k0 < 8192; k0 += 32) {
        // --- A tile: async bf16 copy global -> LDS (no VGPR round-trip)
        asm volatile("global_load_async_to_lds_b128 %0, %1, off"
                     :: "v"(lds_a0), "v"(ga0 + k0) : "memory");
        asm volatile("global_load_async_to_lds_b128 %0, %1, off"
                     :: "v"(lds_a1), "v"(ga1 + k0) : "memory");

        // --- B tile: permuted w2 rows, f32 -> bf16, packed k-pair stores
        {
            const int r  = t >> 4;            // k-pair 0..15
            const int c4 = t & 15;            // float4 column 0..15
            const int f1a = k0 + 2 * r;
            const int f1b = f1a + 1;
            const int r2a = ((f1a & 63) << 7) | (f1a >> 6);
            const int r2b = ((f1b & 63) << 7) | (f1b >> 6);
            const float4 s0 = *(const float4*)(w2 + (size_t)r2a * 1024 + n0 + c4 * 4);
            const float4 s1 = *(const float4*)(w2 + (size_t)r2b * 1024 + n0 + c4 * 4);
            const int f1n = f1a + 32;         // prefetch next K tile's row
            const int r2n = ((f1n & 63) << 7) | ((f1n >> 6) & 127);
            __builtin_prefetch(w2 + (size_t)r2n * 1024 + n0 + c4 * 4, 0, 0);
            const int n = c4 * 4;
            *(unsigned int*)(&Bs[(n + 0) * LDT + 2 * r]) = pack2(s0.x, s1.x);
            *(unsigned int*)(&Bs[(n + 1) * LDT + 2 * r]) = pack2(s0.y, s1.y);
            *(unsigned int*)(&Bs[(n + 2) * LDT + 2 * r]) = pack2(s0.z, s1.z);
            *(unsigned int*)(&Bs[(n + 3) * LDT + 2 * r]) = pack2(s0.w, s1.w);
        }

        asm volatile("s_wait_asynccnt 0x0" ::: "memory");
        __syncthreads();

        FragBF a0, a1;
        const int ar = wr * 32 + lm;
        a0.u[0] = *(const uint4*)(&As[ar * LDT + half * 8]);
        a0.u[1] = *(const uint4*)(&As[ar * LDT + 16 + half * 8]);
        a1.u[0] = *(const uint4*)(&As[(ar + 16) * LDT + half * 8]);
        a1.u[1] = *(const uint4*)(&As[(ar + 16) * LDT + 16 + half * 8]);
#pragma unroll
        for (int nn = 0; nn < 2; ++nn) {
            FragBF b;
            const int br = wc * 32 + nn * 16 + lm;
            b.u[0] = *(const uint4*)(&Bs[br * LDT + half * 8]);
            b.u[1] = *(const uint4*)(&Bs[br * LDT + 16 + half * 8]);
            acc[0][nn] = wmma_bf16(a0, b, acc[0][nn]);
            acc[1][nn] = wmma_bf16(a1, b, acc[1][nn]);
        }
        __syncthreads();
    }

    // --- epilogue: + bias, f32 stores
#pragma unroll
    for (int mi = 0; mi < 2; ++mi) {
        const int brow = m0 + wr * 32 + mi * 16 + half * 8;
#pragma unroll
        for (int nn = 0; nn < 2; ++nn) {
            const int col  = n0 + wc * 32 + nn * 16 + lm;
            const float bv = b2[col];
#pragma unroll
            for (int r = 0; r < 8; ++r) {
                res[(size_t)(brow + r) * 1024 + col] = acc[mi][nn][r] + bv;
            }
        }
    }
}

// ---------------------------------------------------------------------------
// Bit-copy quant_mask (int32) into the head of d_out (tuple output, part 1)
// ---------------------------------------------------------------------------
__global__ void copy_mask_kernel(const int* __restrict__ qm, int* __restrict__ out, int n) {
    const int i = blockIdx.x * blockDim.x + threadIdx.x;
    if (i < n) out[i] = qm[i];
}

extern "C" void kernel_launch(void* const* d_in, const int* in_sizes, int n_in,
                              void* d_out, int out_size, void* d_ws, size_t ws_size,
                              hipStream_t stream) {
    const int*   qmask = (const int*)  d_in[0];  // (2048, 128)
    const float* x     = (const float*)d_in[1];  // (2048, 1024)
    const float* w1    = (const float*)d_in[2];  // (1024, 128, 64)
    const float* w2    = (const float*)d_in[3];  // (64, 128, 1024)
    const float* b2    = (const float*)d_in[4];  // (1024,)

    unsigned short* act = (unsigned short*)d_ws; // 2048*8192 bf16 = 32 MB scratch

    const int nmask = 2048 * 128;
    copy_mask_kernel<<<(nmask + 255) / 256, 256, 0, stream>>>(qmask, (int*)d_out, nmask);

    gemm1_mask_kernel<<<dim3(8192 / 128, 2048 / 128), 256, 0, stream>>>(x, w1, qmask, act);

    float* res = (float*)d_out + nmask;
    gemm2_bias_kernel<<<dim3(1024 / 64, 2048 / 128), 256, 0, stream>>>(act, w2, b2, res);
}
